// BertCRFModel_42717744726634
// MI455X (gfx1250) — matrix operand order repaired and checked by reading, not compiled, not run on hardware
//
#include <hip/hip_runtime.h>
#include <hip/hip_bf16.h>

typedef __attribute__((ext_vector_type(16))) __bf16 v16bf;
typedef __attribute__((ext_vector_type(8)))  float  v8f;

#define Bc 64
#define Sc 512
#define Hc 768
#define Lc 9
#define NPAD 16
#define EM_ELEMS (Bc * Sc * Lc)

// ---------------------------------------------------------------------------
// Kernel 1: emissions = hidden @ W + b  via bf16 hi/lo split WMMA (f32-accurate)
// One wave per 16-row M tile; K swept in 32-wide chunks (24 iterations).
// ---------------------------------------------------------------------------
__global__ __launch_bounds__(256) void emis_gemm_kernel(
    const float* __restrict__ hidden, const float* __restrict__ W,
    const float* __restrict__ bias, float* __restrict__ em) {
  __shared__ __bf16 Wh[Hc * NPAD];
  __shared__ __bf16 Wl[Hc * NPAD];
  // Split W (768x9 f32, padded to N=16) into bf16 hi/lo in LDS.
  for (int idx = threadIdx.x; idx < Hc * NPAD; idx += blockDim.x) {
    int k = idx >> 4, n = idx & 15;
    float w = (n < Lc) ? W[k * Lc + n] : 0.0f;
    __bf16 h = (__bf16)w;
    Wh[idx] = h;
    Wl[idx] = (__bf16)(w - (float)h);
  }
  __syncthreads();

  const int lane = threadIdx.x & 31;
  const int wid  = threadIdx.x >> 5;
  const int tile = blockIdx.x * 8 + wid;   // 16-row tile index over M = B*S
  const int half = lane >> 4;              // 0: lanes 0-15, 1: lanes 16-31
  const int col  = lane & 15;              // output column (N) for C layout
  const float* arow = hidden + (size_t)(tile * 16 + (lane & 15)) * Hc;
  const float bn = (col < Lc) ? bias[col] : 0.0f;

  v8f c = {};
  #pragma unroll 4
  for (int kk = 0; kk < Hc / 32; ++kk) {
    const int kbase = kk * 32;
    // A fragment: lane holds K = kbase + 8*half + [0..7] and +16 (ISA 7.12.2)
    const int k0 = kbase + half * 8;
    float4 f0 = *(const float4*)(arow + k0);
    float4 f1 = *(const float4*)(arow + k0 + 4);
    float4 f2 = *(const float4*)(arow + k0 + 16);
    float4 f3 = *(const float4*)(arow + k0 + 20);
    float av[16] = {f0.x, f0.y, f0.z, f0.w, f1.x, f1.y, f1.z, f1.w,
                    f2.x, f2.y, f2.z, f2.w, f3.x, f3.y, f3.z, f3.w};
    v16bf ah, al;
    #pragma unroll
    for (int e = 0; e < 16; ++e) {
      __bf16 h = (__bf16)av[e];
      ah[e] = h;
      al[e] = (__bf16)(av[e] - (float)h);
    }
    // B fragment: lane holds column N = lane%16, K = kbase + 8*half + [0..7] and +16 (same as A)
    v16bf bh, bl;
    #pragma unroll
    for (int e = 0; e < 16; ++e) {
      const int krow = k0 + (e & 7) + ((e >> 3) << 4);
      bh[e] = Wh[krow * NPAD + col];
      bl[e] = Wl[krow * NPAD + col];
    }
    // D = Ah*Bh + Ah*Bl + Al*Bh  (error term Al*Bl ~ 2^-16, dropped)
    c = __builtin_amdgcn_wmma_f32_16x16x32_bf16(false, ah, false, bh, (short)0, c, false, false);
    c = __builtin_amdgcn_wmma_f32_16x16x32_bf16(false, ah, false, bl, (short)0, c, false, false);
    c = __builtin_amdgcn_wmma_f32_16x16x32_bf16(false, al, false, bh, (short)0, c, false, false);
  }
  if (col < Lc) {
    #pragma unroll
    for (int r = 0; r < 8; ++r) {
      int row = tile * 16 + r + half * 8;   // C layout: VGPR r -> M = r + 8*half
      em[row * Lc + col] = c[r] + bn;
    }
  }
}

// ---------------------------------------------------------------------------
// Kernel 2: CRF NLL per batch. One wave per batch; lane j (<9) owns alpha[j].
// ---------------------------------------------------------------------------
__global__ __launch_bounds__(32) void crf_kernel(
    const float* __restrict__ em, const float* __restrict__ startT,
    const float* __restrict__ trans, const float* __restrict__ endT,
    const int* __restrict__ labels, const int* __restrict__ lengths,
    float* __restrict__ partial) {
  const int b = blockIdx.x;
  const int lane = threadIdx.x;
  int len = lengths[b]; if (len < 1) len = 1;
  const float* eb = em + (size_t)b * Sc * Lc;
  const int*   lb = labels + (size_t)b * Sc;
  const float NEG = -1.0e30f;

  float tr[Lc];
  #pragma unroll
  for (int i = 0; i < Lc; ++i) tr[i] = (lane < Lc) ? trans[i * Lc + lane] : NEG;

  float alpha = (lane < Lc) ? (startT[lane] + eb[lane]) : NEG;
  for (int t = 1; t < Sc; ++t) {
    if (t >= len) break;                 // uniform per block: early exit
    float e_t = (lane < Lc) ? eb[t * Lc + lane] : 0.0f;
    float terms[Lc];
    float m = NEG;
    #pragma unroll
    for (int i = 0; i < Lc; ++i) {
      float ai = __shfl(alpha, i, 32);
      terms[i] = ai + tr[i];
      m = fmaxf(m, terms[i]);
    }
    float s = 0.0f;
    #pragma unroll
    for (int i = 0; i < Lc; ++i) s += __expf(terms[i] - m);
    float nxt = e_t + m + __logf(s);
    if (lane < Lc) alpha = nxt;
  }

  // denominator = logsumexp_j(alpha[j] + end[j])
  float x = (lane < Lc) ? (alpha + endT[lane]) : NEG;
  float m = x;
  #pragma unroll
  for (int off = 16; off > 0; off >>= 1) m = fmaxf(m, __shfl_xor(m, off, 32));
  float s = (lane < Lc) ? __expf(x - m) : 0.0f;
  #pragma unroll
  for (int off = 16; off > 0; off >>= 1) s += __shfl_xor(s, off, 32);
  float den = m + __logf(s);

  // numerator: parallel strided sum over time steps
  float num = 0.0f;
  for (int t = 1 + lane; t < len; t += 32) {
    int tp = lb[t - 1], tc = lb[t];
    num += trans[tp * Lc + tc] + eb[t * Lc + tc];
  }
  if (lane == 0) {
    int t0 = lb[0];
    num += startT[t0] + eb[t0] + endT[lb[len - 1]];
  }
  #pragma unroll
  for (int off = 16; off > 0; off >>= 1) num += __shfl_xor(num, off, 32);
  if (lane == 0) partial[b] = den - num;
}

// ---------------------------------------------------------------------------
// Kernel 3: deterministic fixed-order final reduction of 64 partials.
// ---------------------------------------------------------------------------
__global__ __launch_bounds__(32) void final_reduce(
    const float* __restrict__ partial, float* __restrict__ out) {
  int lane = threadIdx.x;
  float v = partial[lane] + partial[lane + 32];
  #pragma unroll
  for (int off = 16; off > 0; off >>= 1) v += __shfl_xor(v, off, 32);
  if (lane == 0) out[0] = v;
}

extern "C" void kernel_launch(void* const* d_in, const int* in_sizes, int n_in,
                              void* d_out, int out_size, void* d_ws, size_t ws_size,
                              hipStream_t stream) {
  (void)in_sizes; (void)n_in; (void)out_size; (void)ws_size;
  const float* hidden = (const float*)d_in[0];
  const float* W      = (const float*)d_in[1];
  const float* bias   = (const float*)d_in[2];
  const float* startT = (const float*)d_in[3];
  const float* trans  = (const float*)d_in[4];
  const float* endT   = (const float*)d_in[5];
  const int*   labels = (const int*)d_in[6];
  const int*   lengths= (const int*)d_in[7];

  float* em      = (float*)d_ws;
  float* partial = em + EM_ELEMS;
  float* out     = (float*)d_out;

  const int tiles = (Bc * Sc) / 16;              // 2048 M-tiles
  emis_gemm_kernel<<<tiles / 8, 256, 0, stream>>>(hidden, W, bias, em);
  crf_kernel<<<Bc, 32, 0, stream>>>(em, startT, trans, endT, labels, lengths, partial);
  final_reduce<<<1, 32, 0, stream>>>(partial, out);
}
